// DockPointNet_72413148610698
// MI455X (gfx1250) — compile-verified, matrix operand order checked by reading, not run on hardware
//
#include <hip/hip_runtime.h>
#include <math.h>

// ---------------- problem constants (from reference) ----------------
#define BB   8
#define NPTS 4096
#define NN   (BB*NPTS)          // 32768
#define SS   2048               // fps samples per graph
#define KK   20                 // max neighbors
#define EE   (BB*SS*KK)         // 327680 edges
#define R2C  0.04f              // radius^2
#define EPSV 1e-5f

typedef _Float16 h16 __attribute__((ext_vector_type(16)));
typedef _Float16 h8  __attribute__((ext_vector_type(8)));
typedef float    f8  __attribute__((ext_vector_type(8)));

// ---------------- utility: zero a u32 region ----------------
__global__ void zero_kernel(unsigned* p, size_t cnt) {
    size_t i = (size_t)blockIdx.x * blockDim.x + threadIdx.x;
    for (; i < cnt; i += (size_t)gridDim.x * blockDim.x) p[i] = 0u;
}

// ---------------- farthest point sampling (one block per graph) ----------------
__global__ void fps_kernel(const float* __restrict__ X, int dim, int stride,
                           int* __restrict__ idx) {
    __shared__ float mind[NPTS];
    __shared__ float rv[256];
    __shared__ int   ri[256];
    __shared__ float lastv[64];
    __shared__ int   s_last;
    const int b = blockIdx.x, tid = threadIdx.x;
    const float* Xb = X + (size_t)b * NPTS * stride;
    for (int j = tid; j < NPTS; j += 256) mind[j] = 3.4e38f;
    if (tid == 0) { idx[b * SS] = 0; s_last = 0; }
    __syncthreads();
    for (int i = 1; i < SS; ++i) {
        const int last = s_last;
        for (int d = tid; d < dim; d += 256) lastv[d] = Xb[(size_t)last * stride + d];
        __syncthreads();
        float bv = -1.0f; int bi = NPTS;
        for (int j = tid; j < NPTS; j += 256) {
            float acc = 0.0f;
            for (int d = 0; d < dim; ++d) {
                float df = Xb[(size_t)j * stride + d] - lastv[d];
                acc += df * df;
            }
            float m = fminf(mind[j], acc);
            mind[j] = m;
            if (m > bv || (m == bv && j < bi)) { bv = m; bi = j; }
        }
        rv[tid] = bv; ri[tid] = bi;
        __syncthreads();
        for (int s = 128; s > 0; s >>= 1) {
            if (tid < s) {
                float ov = rv[tid + s]; int oi = ri[tid + s];
                if (ov > rv[tid] || (ov == rv[tid] && oi < ri[tid])) { rv[tid] = ov; ri[tid] = oi; }
            }
            __syncthreads();
        }
        if (tid == 0) { idx[b * SS + i] = ri[0]; s_last = ri[0]; }
        __syncthreads();
    }
}

// ---------------- radius kNN: one thread per query, insertion top-K ----------------
__global__ void knn_kernel(const float* __restrict__ X, int dim, int stride,
                           const int* __restrict__ idx,
                           int* __restrict__ cols, int* __restrict__ valid) {
    const int q = blockIdx.x * blockDim.x + threadIdx.x;
    if (q >= BB * SS) return;
    const int b = q / SS;
    const int lq = idx[q];
    const float* Xb = X + (size_t)b * NPTS * stride;
    float qv[64];
    for (int d = 0; d < dim; ++d) qv[d] = Xb[(size_t)lq * stride + d];
    float bd[KK]; int bc[KK]; int cnt = 0;
    for (int j = 0; j < NPTS; ++j) {
        float acc = 0.0f;
        for (int d = 0; d < dim; ++d) {
            float df = Xb[(size_t)j * stride + d] - qv[d];
            acc += df * df;
        }
        if (acc <= R2C) {
            if (cnt < KK) {
                int p = cnt++;
                while (p > 0 && bd[p - 1] > acc) { bd[p] = bd[p - 1]; bc[p] = bc[p - 1]; --p; }
                bd[p] = acc; bc[p] = j;
            } else if (acc < bd[KK - 1]) {
                int p = KK - 1;
                while (p > 0 && bd[p - 1] > acc) { bd[p] = bd[p - 1]; bc[p] = bc[p - 1]; --p; }
                bd[p] = acc; bc[p] = j;
            }
        }
    }
    for (int t = 0; t < KK; ++t) {
        valid[q * KK + t] = (t < cnt) ? 1 : 0;
        cols[q * KK + t]  = b * NPTS + ((t < cnt) ? bc[t] : 0);
    }
}

// ---------------- edge gather for conv1: feat = [x_i, x_j - x_i], pad 6->32 ----------------
__global__ void gather1_kernel(const float* __restrict__ pos, const int* __restrict__ cols,
                               _Float16* __restrict__ out) {
    const int e = blockIdx.x * blockDim.x + threadIdx.x;
    if (e >= EE) return;
    const int q = e / KK;          // row_g == q (reference quirk: x_j = x[row])
    const int c = cols[e];
    float xi0 = pos[c * 3 + 0], xi1 = pos[c * 3 + 1], xi2 = pos[c * 3 + 2];
    float xj0 = pos[q * 3 + 0], xj1 = pos[q * 3 + 1], xj2 = pos[q * 3 + 2];
    _Float16* o = out + (size_t)e * 32;
    o[0] = (_Float16)xi0; o[1] = (_Float16)xi1; o[2] = (_Float16)xi2;
    o[3] = (_Float16)(xj0 - xi0); o[4] = (_Float16)(xj1 - xi1); o[5] = (_Float16)(xj2 - xi2);
    for (int f = 6; f < 32; ++f) o[f] = (_Float16)0.0f;
}

// ---------------- edge gather for conv2 (64-dim features) ----------------
__global__ void gather2_kernel(const float* __restrict__ x1, const int* __restrict__ cols,
                               _Float16* __restrict__ out) {
    const int e = blockIdx.x * blockDim.x + threadIdx.x;
    if (e >= EE) return;
    const int q = e / KK;
    const int c = cols[e];
    const float* xi = x1 + (size_t)c * 64;
    const float* xj = x1 + (size_t)q * 64;
    _Float16* o = out + (size_t)e * 128;
    for (int f = 0; f < 64; ++f) {
        float a = xi[f];
        o[f]      = (_Float16)a;
        o[64 + f] = (_Float16)(xj[f] - a);
    }
}

// ---------------- pack fp32 weights [din,dout] -> f16 WMMA B-fragment order ----------
// layout: wp[ ((kb/32)*tilesN + tn)*512 + lane*16 + j ], so each lane's 16-half
// fragment is contiguous (two 16-byte chunks -> global_load_b128 pairs in GEMM).
__global__ void packw_frag_kernel(const float* __restrict__ w, int din, int dout, int kpad,
                                  _Float16* __restrict__ wp) {
    const int i = blockIdx.x * blockDim.x + threadIdx.x;
    const int tot = (kpad >> 5) * (dout >> 4) * 512;
    if (i >= tot) return;
    const int j    = i & 15;
    const int lane = (i >> 4) & 31;
    const int frag = i >> 9;
    const int tilesN = dout >> 4;
    const int kbi = frag / tilesN, tn = frag % tilesN;
    const int half = lane >> 4, mr = lane & 15;
    const int v = j >> 1, e = j & 1;
    const int k = kbi * 32 + ((v & 3) << 1) + e + (half << 3) + ((v >> 2) << 4);
    const int n = tn * 16 + mr;
    wp[i] = (k < din) ? (_Float16)w[k * dout + n] : (_Float16)0.0f;
}

__device__ __forceinline__ h16 cat8(h8 lo, h8 hi) {
    return __builtin_shufflevector(lo, hi, 0, 1, 2, 3, 4, 5, 6, 7,
                                          8, 9, 10, 11, 12, 13, 14, 15);
}

// ---------------- WMMA GEMM + bias + ReLU ----------------
// One wave per 16x32 output tile (two 16x16 N-tiles sharing the A fragment).
// KD compile-time -> fully unrolled k-loop, back-to-back v_wmma.
// Tile guard is wave-uniform so EXEC is all-ones at every v_wmma.
template <int KD>
__global__ void gemm_wmma_kernel(const _Float16* __restrict__ A, const _Float16* __restrict__ Wf,
                                 const float* __restrict__ bias, _Float16* __restrict__ out,
                                 int M, int Nc) {
    const int tilesN  = Nc >> 4;
    const int tilesN2 = Nc >> 5;
    const int tiles   = (M >> 4) * tilesN2;
    const int wave    = (blockIdx.x * blockDim.x + threadIdx.x) >> 5;
    if (wave >= tiles) return;                         // uniform per wave
    const int lane = threadIdx.x & 31;
    const int tm = wave / tilesN2, tn2 = wave % tilesN2;
    const int half = lane >> 4, mr = lane & 15;
    const _Float16* Arow = A + (size_t)(tm * 16 + mr) * KD + half * 8;
    const _Float16* Wl   = Wf + (size_t)lane * 16;
    f8 acc0 = {0.f, 0.f, 0.f, 0.f, 0.f, 0.f, 0.f, 0.f};
    f8 acc1 = {0.f, 0.f, 0.f, 0.f, 0.f, 0.f, 0.f, 0.f};
#pragma unroll
    for (int kb = 0; kb < KD; kb += 32) {
        const h16 a = cat8(*(const h8*)(Arow + kb), *(const h8*)(Arow + kb + 16));
        const _Float16* wb = Wl + ((size_t)(kb >> 5) * tilesN + 2 * tn2) * 512;
        const h16 b0 = cat8(*(const h8*)(wb),        *(const h8*)(wb + 8));
        const h16 b1 = cat8(*(const h8*)(wb + 512),  *(const h8*)(wb + 520));
        acc0 = __builtin_amdgcn_wmma_f32_16x16x32_f16(false, a, false, b0, (short)0, acc0,
                                                      false, false);
        acc1 = __builtin_amdgcn_wmma_f32_16x16x32_f16(false, a, false, b1, (short)0, acc1,
                                                      false, false);
    }
    const int n0 = tn2 * 32 + mr;
    const int n1 = n0 + 16;
    const float bv0 = bias ? bias[n0] : 0.0f;
    const float bv1 = bias ? bias[n1] : 0.0f;
#pragma unroll
    for (int r = 0; r < 8; ++r) {
        const int m = tm * 16 + r + half * 8;
        float v0 = acc0[r] + bv0; v0 = v0 > 0.0f ? v0 : 0.0f;   // ReLU precedes BN
        float v1 = acc1[r] + bv1; v1 = v1 > 0.0f ? v1 : 0.0f;
        out[(size_t)m * Nc + n0] = (_Float16)v0;
        out[(size_t)m * Nc + n1] = (_Float16)v1;
    }
}

// ---------------- BN statistics (masked): LDS partial sums + global atomics ----------------
__global__ void bn_stats_kernel(const _Float16* __restrict__ Y, const int* __restrict__ valid,
                                int M, int F, float* __restrict__ stats) {
    __shared__ float ls[2048];       // [0,F): sum, [F,2F): sumsq
    __shared__ float lcnt;
    const int tid = threadIdx.x;
    for (int f = tid; f < 2 * F; f += blockDim.x) ls[f] = 0.0f;
    if (tid == 0) lcnt = 0.0f;
    __syncthreads();
    float my = 0.0f;
    for (int r = blockIdx.x * blockDim.x + tid; r < M; r += gridDim.x * blockDim.x) {
        if (valid && !valid[r]) continue;
        my += 1.0f;
        const _Float16* row = Y + (size_t)r * F;
        for (int f0 = 0; f0 < F; ++f0) {
            int f = f0 + tid; if (f >= F) f -= F;    // stagger to spread LDS atomics
            const float v = (float)row[f];
            atomicAdd(&ls[f], v);
            atomicAdd(&ls[F + f], v * v);
        }
    }
    atomicAdd(&lcnt, my);
    __syncthreads();
    for (int f = tid; f < 2 * F; f += blockDim.x)
        if (ls[f] != 0.0f) atomicAdd(&stats[f], ls[f]);
    if (tid == 0 && lcnt != 0.0f) atomicAdd(&stats[2 * F], lcnt);
}

// stats layout: [0,F)=sum [F,2F)=sumsq [2F]=cnt [2F+8,3F+8)=scale [3F+8,4F+8)=shift
__global__ void bn_finalize_kernel(const float* __restrict__ g, const float* __restrict__ be,
                                   int F, float* __restrict__ stats) {
    const int f = blockIdx.x * blockDim.x + threadIdx.x;
    if (f >= F) return;
    float cnt = stats[2 * F]; cnt = cnt < 1.0f ? 1.0f : cnt;
    const float m = stats[f] / cnt;
    float v = stats[F + f] / cnt - m * m; if (v < 0.0f) v = 0.0f;
    const float sc = g[f] * rsqrtf(v + EPSV);
    stats[2 * F + 8 + f] = sc;
    stats[3 * F + 8 + f] = be[f] - m * sc;
}

__global__ void bn_apply_kernel(const _Float16* __restrict__ in, _Float16* __restrict__ out,
                                int M, int F, const float* __restrict__ stats) {
    size_t i = (size_t)blockIdx.x * blockDim.x + threadIdx.x;
    const size_t tot = (size_t)M * F;
    for (; i < tot; i += (size_t)gridDim.x * blockDim.x) {
        const int f = (int)(i % F);
        const float v = (float)in[i] * stats[2 * F + 8 + f] + stats[3 * F + 8 + f];
        out[i] = (_Float16)v;
    }
}

// ---------------- scatter-max via order-preserving u32 encoding ----------------
__device__ __forceinline__ unsigned enc_f32(float f) {
    unsigned u = __float_as_uint(f);
    return (u & 0x80000000u) ? ~u : (u | 0x80000000u);
}
__global__ void scatter_kernel(const _Float16* __restrict__ Y, const int* __restrict__ valid,
                               const int* __restrict__ cols, unsigned* __restrict__ agg, int F) {
    size_t i = (size_t)blockIdx.x * blockDim.x + threadIdx.x;
    const size_t tot = (size_t)EE * F;
    for (; i < tot; i += (size_t)gridDim.x * blockDim.x) {
        const int e = (int)(i / F), f = (int)(i % F);
        if (!valid[e]) continue;
        atomicMax(&agg[(size_t)cols[e] * F + f], enc_f32((float)Y[i]));
    }
}
__device__ __forceinline__ float dec_f32(unsigned u) {
    if (u == 0u) return 0.0f;                        // empty segment -> 0 (torch_scatter)
    const unsigned b = (u & 0x80000000u) ? (u ^ 0x80000000u) : ~u;
    return __uint_as_float(b);
}
__global__ void decode1_kernel(const unsigned* __restrict__ agg, float* __restrict__ x1,
                               _Float16* __restrict__ linA) {
    const int i = blockIdx.x * blockDim.x + threadIdx.x;
    if (i >= NN * 64) return;
    const int n = i >> 6, f = i & 63;
    const float v = dec_f32(agg[i]);
    x1[i] = v;
    linA[(size_t)n * 192 + f] = (_Float16)v;
}
__global__ void decode2_kernel(const unsigned* __restrict__ agg, _Float16* __restrict__ linA) {
    const int i = blockIdx.x * blockDim.x + threadIdx.x;
    if (i >= NN * 128) return;
    const int n = i >> 7, f = i & 127;
    linA[(size_t)n * 192 + 64 + f] = (_Float16)dec_f32(agg[i]);
}

// ---------------- per-graph max pool over 1024 features ----------------
__global__ void pool_kernel(const _Float16* __restrict__ h, float* __restrict__ pooled) {
    const int i = blockIdx.x * blockDim.x + threadIdx.x;
    if (i >= BB * 1024) return;
    const int b = i >> 10, f = i & 1023;
    float mx = -3.4e38f;
    const _Float16* base = h + (size_t)b * NPTS * 1024 + f;
    for (int n = 0; n < NPTS; ++n) mx = fmaxf(mx, (float)base[(size_t)n * 1024]);
    pooled[i] = mx;
}

// ---------------- small dense layers (M=8): scalar path ----------------
__global__ void dense_relu_kernel(const float* __restrict__ in, const float* __restrict__ w,
                                  const float* __restrict__ bias, float* __restrict__ out,
                                  int Fi, int Fo) {
    const int i = blockIdx.x * blockDim.x + threadIdx.x;
    if (i >= BB * Fo) return;
    const int b = i / Fo, o = i % Fo;
    float acc = bias[o];
    const float* r = in + (size_t)b * Fi;
    for (int k = 0; k < Fi; ++k) acc += r[k] * w[(size_t)k * Fo + o];
    out[i] = acc > 0.0f ? acc : 0.0f;
}
__global__ void bn_small_kernel(float* __restrict__ x, int F, const float* __restrict__ g,
                                const float* __restrict__ be) {
    const int f = blockIdx.x * blockDim.x + threadIdx.x;
    if (f >= F) return;
    float s = 0.0f, ss = 0.0f;
    for (int b = 0; b < BB; ++b) { const float v = x[b * F + f]; s += v; ss += v * v; }
    const float m = s / BB;
    float var = ss / BB - m * m; if (var < 0.0f) var = 0.0f;
    const float sc = g[f] * rsqrtf(var + EPSV), sh = be[f] - m * sc;
    for (int b = 0; b < BB; ++b) x[b * F + f] = x[b * F + f] * sc + sh;
}
__global__ void head_kernel(const float* __restrict__ x, const float* __restrict__ w,
                            const float* __restrict__ bias, float* __restrict__ out) {
    const int b = threadIdx.x;
    if (b >= BB) return;
    float z0 = bias[0], z1 = bias[1];
    for (int k = 0; k < 256; ++k) {
        const float v = x[b * 256 + k];
        z0 += v * w[k * 2 + 0];
        z1 += v * w[k * 2 + 1];
    }
    const float mx = fmaxf(z0, z1);
    const float l = mx + logf(expf(z0 - mx) + expf(z1 - mx));
    out[b * 2 + 0] = z0 - l;
    out[b * 2 + 1] = z1 - l;
}

// =====================================================================
extern "C" void kernel_launch(void* const* d_in, const int* in_sizes, int n_in,
                              void* d_out, int out_size, void* d_ws, size_t ws_size,
                              hipStream_t stream) {
    (void)in_sizes; (void)n_in; (void)out_size; (void)ws_size;
    const float* pos = (const float*)d_in[0];
    // d_in[1] = batch (implicit in layout, unused)
    const float *c1l0_w = (const float*)d_in[2],  *c1l0_b = (const float*)d_in[3],
                *c1l0_g = (const float*)d_in[4],  *c1l0_be = (const float*)d_in[5];
    const float *c1l1_w = (const float*)d_in[6],  *c1l1_b = (const float*)d_in[7],
                *c1l1_g = (const float*)d_in[8],  *c1l1_be = (const float*)d_in[9];
    const float *c1l2_w = (const float*)d_in[10], *c1l2_b = (const float*)d_in[11],
                *c1l2_g = (const float*)d_in[12], *c1l2_be = (const float*)d_in[13];
    const float *c2l0_w = (const float*)d_in[14], *c2l0_b = (const float*)d_in[15],
                *c2l0_g = (const float*)d_in[16], *c2l0_be = (const float*)d_in[17];
    const float *lin1_w = (const float*)d_in[18], *lin1_b = (const float*)d_in[19],
                *lin1_g = (const float*)d_in[20], *lin1_be = (const float*)d_in[21];
    const float *ma_w = (const float*)d_in[22], *ma_b = (const float*)d_in[23],
                *ma_g = (const float*)d_in[24], *ma_be = (const float*)d_in[25];
    const float *mb_w = (const float*)d_in[26], *mb_b = (const float*)d_in[27],
                *mb_g = (const float*)d_in[28], *mb_be = (const float*)d_in[29];
    const float *head_w = (const float*)d_in[30], *head_b = (const float*)d_in[31];
    float* out = (float*)d_out;

    // ---- workspace carve-out (~209 MB) ----
    char* ws = (char*)d_ws;
    size_t off = 0;
    auto carve = [&](size_t bytes) -> char* {
        char* p = ws + off;
        off = (off + bytes + 255) & ~(size_t)255;
        return p;
    };
    int*       idx   = (int*)carve((size_t)BB * SS * 4);
    int*       cols  = (int*)carve((size_t)EE * 4);
    int*       valid = (int*)carve((size_t)EE * 4);
    float*     x1f   = (float*)carve((size_t)NN * 64 * 4);
    _Float16*  linA  = (_Float16*)carve((size_t)NN * 192 * 2);
    unsigned*  agg   = (unsigned*)carve((size_t)NN * 128 * 4);
    _Float16*  ebufA = (_Float16*)carve((size_t)EE * 128 * 2);
    _Float16*  ebufB = (_Float16*)carve((size_t)EE * 128 * 2);  // also lin1 out (67MB<84MB)
    _Float16*  wpad  = (_Float16*)carve((size_t)192 * 1024 * 2);
    float*     stats = (float*)carve((size_t)(4 * 1024 + 8) * 4);
    float*     pooled= (float*)carve((size_t)BB * 1024 * 4);
    float*     fc1   = (float*)carve((size_t)BB * 512 * 4);
    float*     fc2   = (float*)carve((size_t)BB * 256 * 4);

    const dim3 blk(256);
    auto g1 = [](size_t n) { return dim3((unsigned)((n + 255) / 256)); };

    // launch the right fully-unrolled GEMM instantiation
    auto launch_gemm = [&](const _Float16* Ain, int kd, const float* bias,
                           _Float16* outb, int M, int Nc) {
        const size_t waves = (size_t)(M / 16) * (Nc / 32);
        const dim3 grid = g1(waves * 32);
        switch (kd) {
            case 32:  gemm_wmma_kernel<32><<<grid, blk, 0, stream>>>(Ain, wpad, bias, outb, M, Nc); break;
            case 64:  gemm_wmma_kernel<64><<<grid, blk, 0, stream>>>(Ain, wpad, bias, outb, M, Nc); break;
            case 128: gemm_wmma_kernel<128><<<grid, blk, 0, stream>>>(Ain, wpad, bias, outb, M, Nc); break;
            default:  gemm_wmma_kernel<192><<<grid, blk, 0, stream>>>(Ain, wpad, bias, outb, M, Nc); break;
        }
    };

    // one Lin+ReLU+BN(masked) edge-MLP layer: gemm into outb, normalize into nrm
    auto edge_layer = [&](const _Float16* in, int kd, _Float16* outb,
                          const float* w, int din, const float* bias,
                          const float* g, const float* be, int F, _Float16* nrm) {
        packw_frag_kernel<<<g1((size_t)kd * F), blk, 0, stream>>>(w, din, F, kd, wpad);
        launch_gemm(in, kd, bias, outb, EE, F);
        zero_kernel<<<g1(2 * F + 8), blk, 0, stream>>>((unsigned*)stats, (size_t)(2 * F + 8));
        bn_stats_kernel<<<dim3(512), blk, 0, stream>>>(outb, valid, EE, F, stats);
        bn_finalize_kernel<<<g1(F), blk, 0, stream>>>(g, be, F, stats);
        bn_apply_kernel<<<g1((size_t)EE * F), blk, 0, stream>>>(outb, nrm, EE, F, stats);
    };

    // ================= conv1 (on positions, dim 3) =================
    fps_kernel<<<dim3(BB), blk, 0, stream>>>(pos, 3, 3, idx);
    knn_kernel<<<g1(BB * SS), blk, 0, stream>>>(pos, 3, 3, idx, cols, valid);
    gather1_kernel<<<g1(EE), blk, 0, stream>>>(pos, cols, ebufA);
    edge_layer(ebufA, 32, ebufB, c1l0_w, 6,  c1l0_b, c1l0_g, c1l0_be, 64, ebufA);
    edge_layer(ebufA, 64, ebufB, c1l1_w, 64, c1l1_b, c1l1_g, c1l1_be, 64, ebufA);
    edge_layer(ebufA, 64, ebufB, c1l2_w, 64, c1l2_b, c1l2_g, c1l2_be, 64, ebufA);
    zero_kernel<<<g1((size_t)NN * 64), blk, 0, stream>>>(agg, (size_t)NN * 64);
    scatter_kernel<<<g1((size_t)EE * 64), blk, 0, stream>>>(ebufA, valid, cols, agg, 64);
    decode1_kernel<<<g1((size_t)NN * 64), blk, 0, stream>>>(agg, x1f, linA);

    // ================= conv2 (on x1 features, dim 64) =================
    fps_kernel<<<dim3(BB), blk, 0, stream>>>(x1f, 64, 64, idx);
    knn_kernel<<<g1(BB * SS), blk, 0, stream>>>(x1f, 64, 64, idx, cols, valid);
    gather2_kernel<<<g1(EE), blk, 0, stream>>>(x1f, cols, ebufA);
    edge_layer(ebufA, 128, ebufB, c2l0_w, 128, c2l0_b, c2l0_g, c2l0_be, 128, ebufA);
    zero_kernel<<<g1((size_t)NN * 128), blk, 0, stream>>>(agg, (size_t)NN * 128);
    scatter_kernel<<<g1((size_t)EE * 128), blk, 0, stream>>>(ebufA, valid, cols, agg, 128);
    decode2_kernel<<<g1((size_t)NN * 128), blk, 0, stream>>>(agg, linA);

    // ================= lin1: [N,192] @ [192,1024] + BN + max-pool =================
    packw_frag_kernel<<<g1((size_t)192 * 1024), blk, 0, stream>>>(lin1_w, 192, 1024, 192, wpad);
    launch_gemm(linA, 192, lin1_b, ebufB, NN, 1024);
    zero_kernel<<<g1(2 * 1024 + 8), blk, 0, stream>>>((unsigned*)stats, (size_t)(2 * 1024 + 8));
    bn_stats_kernel<<<dim3(512), blk, 0, stream>>>(ebufB, (const int*)nullptr, NN, 1024, stats);
    bn_finalize_kernel<<<g1(1024), blk, 0, stream>>>(lin1_g, lin1_be, 1024, stats);
    bn_apply_kernel<<<g1((size_t)NN * 1024), blk, 0, stream>>>(ebufB, ebufB, NN, 1024, stats);
    pool_kernel<<<g1((size_t)BB * 1024), blk, 0, stream>>>(ebufB, pooled);

    // ================= small FC head (M=8, scalar) =================
    dense_relu_kernel<<<g1((size_t)BB * 512), blk, 0, stream>>>(pooled, ma_w, ma_b, fc1, 1024, 512);
    bn_small_kernel<<<g1(512), blk, 0, stream>>>(fc1, 512, ma_g, ma_be);
    dense_relu_kernel<<<g1((size_t)BB * 256), blk, 0, stream>>>(fc1, mb_w, mb_b, fc2, 512, 256);
    bn_small_kernel<<<g1(256), blk, 0, stream>>>(fc2, 256, mb_g, mb_be);
    head_kernel<<<dim3(1), dim3(32), 0, stream>>>(fc2, head_w, head_b, out);
}
